// EntNetQA_60756607369818
// MI455X (gfx1250) — matrix operand order repaired and checked by reading, not compiled, not run on hardware
//
#include <hip/hip_runtime.h>
#include <math.h>

typedef __attribute__((ext_vector_type(16))) _Float16 v16h;
typedef __attribute__((ext_vector_type(8)))  _Float16 v8h;
typedef __attribute__((ext_vector_type(8)))  float    v8f;
typedef __attribute__((ext_vector_type(4)))  float    v4f;
typedef __attribute__((ext_vector_type(4)))  _Float16 v4h;
typedef __attribute__((ext_vector_type(4)))  unsigned u32x4;
typedef __attribute__((ext_vector_type(8)))  int      i32x8;
typedef __attribute__((ext_vector_type(4)))  int      i32x4;

#define EMBED 256
#define NB    20          // NUM_BLOCKS
#define QQ    20          // queries per batch
#define BATCH 64
#define RN    10
#define LEN   60
#define SEQ   20
#define ROUT  20000
#define KEY0  20000       // VOCAB_TOTAL - NUM_BLOCKS

// ---------------- helpers ----------------

__device__ __forceinline__ float wave_reduce(float v) {
  v += __shfl_xor(v, 16);
  v += __shfl_xor(v, 8);
  v += __shfl_xor(v, 4);
  v += __shfl_xor(v, 2);
  v += __shfl_xor(v, 1);
  return v;
}

// A fragment: 16x32 (MxK) f16, contiguous k order (k = 16*(lane/16) + s), matching B.
__device__ __forceinline__ v16h load_a_frag(const _Float16* lds, int lda,
                                            int m0, int k0, int lane) {
  int m  = m0 + (lane & 15);
  int kb = k0 + ((lane >> 4) << 4);
  return *(const v16h*)(lds + m * lda + kb);
}

// B fragment: 32x16 (KxN) f16 where B[k][n] = Wmat[n][k] (Wmat row-major).
__device__ __forceinline__ v16h load_b_frag(const _Float16* lds, int ldb,
                                            int n0, int k0, int lane) {
  int n  = n0 + (lane & 15);
  int kb = k0 + ((lane >> 4) << 4);
  return *(const v16h*)(lds + n * ldb + kb);
}

// TDM: async-copy `rows` x 256-half tile (row-major, element stride 256) from
// global f16 into LDS at byte offset `lds_byte`, inserting 32B padding after
// every 512B row so the LDS pitch is 272 halfs (matches the WMMA frag loads).
// D# packing per CDNA5 ISA 8.3/8.4: count=1, type=2(image), data_size=1(2B),
// pad_interval=6 (128 DWORDs = 512B), pad_amount=7 (8 DWORDs = 32B).
// This toolchain's builtin is the 6-arg form:
//   (u32x4 g0, i32x8 g1, i32x4 g2, i32x4 g3, i32x8, i32 cpol)
__device__ __forceinline__ void tdm_load_rows(const _Float16* gsrc,
                                              unsigned lds_byte, int rows) {
  unsigned long long ga = (unsigned long long)gsrc;
  u32x4 g0;
  g0[0] = 1u;                                          // count=1, user mode
  g0[1] = lds_byte;                                    // lds_addr
  g0[2] = (unsigned)ga;                                // global_addr[31:0]
  g0[3] = ((unsigned)(ga >> 32) & 0x1FFFFFFu)          // global_addr[56:32]
          | (2u << 30);                                // type=2
  i32x8 g1;
  g1[0] = (int)((1u << 16)      // data_size = 2 bytes
              | (1u << 20)      // pad_enable
              | (6u << 22)      // pad_interval: 128 DWORDs
              | (7u << 25));    // pad_amount:   8 DWORDs
  g1[1] = (int)(256u << 16);                 // tensor_dim0 = 256  [79:48]
  g1[2] = (int)((unsigned)rows << 16);       // tensor_dim1 = rows [111:80]
  g1[3] = (int)(256u << 16);                 // tile_dim0 = 256    [127:112]
  g1[4] = rows;                              // tile_dim1          [143:128]
  g1[5] = 256;                               // tensor_dim0_stride [207:160]
  g1[6] = 0;
  g1[7] = 0;
  i32x4 z4 = {0, 0, 0, 0};
  i32x8 z8 = {0, 0, 0, 0, 0, 0, 0, 0};
  __builtin_amdgcn_tensor_load_to_lds(g0, g1, z4, z4, z8, 0);
}

// ---------------- embedding bags ----------------

__global__ void hist_kernel(const int* fr, const float* embed,
                            const float* smask, float* hist) {
  int br = blockIdx.x;       // b*RN + r   (640)
  int e  = threadIdx.x;      // 256
  float acc = 0.f;
  for (int l = 0; l < LEN; ++l) {
    int idx = fr[br * LEN + l];
    if (idx != 0) acc += embed[idx * EMBED + e] * smask[l * EMBED + e];
  }
  hist[br * EMBED + e] = acc;
}

__global__ void qenc_kernel(const int* qa, const float* embed,
                            const float* qmask, float* qenc) {
  int bq = blockIdx.x;       // 1280
  int e  = threadIdx.x;
  float acc = 0.f;
  for (int s = 0; s < SEQ; ++s) {
    int idx = qa[bq * SEQ + s];
    if (idx != 0) acc += embed[idx * EMBED + e] * qmask[s * EMBED + e];
  }
  qenc[bq * EMBED + e] = acc;
}

// ---------------- small state-independent precomputes ----------------

__global__ void keyv_kernel(const float* embed, const float* V, float* keyV) {
  __shared__ float kls[EMBED];
  int k = blockIdx.x, f = threadIdx.x;
  kls[f] = embed[(KEY0 + k) * EMBED + f];
  __syncthreads();
  float acc = 0.f;
  for (int e = 0; e < EMBED; ++e) acc += kls[e] * V[f * EMBED + e];
  keyV[k * EMBED + f] = acc;
}

__global__ void xw_kernel(const float* hist, const float* W,
                          const float* bias, float* XW) {
  __shared__ float hls[EMBED];
  int bt = blockIdx.x, f = threadIdx.x;
  hls[f] = hist[bt * EMBED + f];
  __syncthreads();
  float acc = bias[f];
  for (int e = 0; e < EMBED; ++e) acc += hls[e] * W[f * EMBED + e];
  XW[bt * EMBED + f] = acc;
}

__global__ void xk_kernel(const float* hist, const float* embed, float* XK) {
  int bt   = blockIdx.x;
  int lane = threadIdx.x & 31, w = threadIdx.x >> 5;
  for (int k = w; k < NB; k += 8) {
    float p = 0.f;
    for (int j = 0; j < 8; ++j) {
      int e = lane + 32 * j;
      p += hist[bt * EMBED + e] * embed[(KEY0 + k) * EMBED + e];
    }
    p = wave_reduce(p);
    if (lane == 0) XK[bt * NB + k] = p;
  }
}

// f32 -> f16 convert for R (one-shot; makes R TDM-copyable)
__global__ void cvt_r_kernel(const float* R, _Float16* Rh) {
  int i = (blockIdx.x * 256 + threadIdx.x) * 4;
  v4f v = *(const v4f*)(R + i);
  v4h h;
  h[0] = (_Float16)v[0]; h[1] = (_Float16)v[1];
  h[2] = (_Float16)v[2]; h[3] = (_Float16)v[3];
  *(v4h*)(Rh + i) = h;
}

// ---------------- the recurrent scan (state depends only on b!) ----------------

__global__ void scan_kernel(const float* embed, const float* U,
                            const float* prelu_a, const float* hist,
                            const float* XW, const float* XK,
                            const float* keyV, float* stateOut) {
  extern __shared__ char smem[];
  _Float16* Uh  = (_Float16*)smem;                          // 256 x 272 f16
  _Float16* sH  = (_Float16*)(smem + 139264);               // 32 x 272 f16
  float*    sF  = (float*)(smem + 139264 + 17408);          // 32 x 256 f32
  float*    cF  = (float*)((char*)sF + 32768);              // 32 x 256 f32
  float*    kV  = (float*)((char*)cF + 32768);              // 20 x 256 f32
  float*    xls = (float*)((char*)kV + 20480);              // 256
  float*    xwls  = xls + 256;                              // 256
  float*    pals  = xwls + 256;                             // 256
  float*    gate  = pals + 256;                             // 32
  float*    norms = gate + 32;                              // 32

  int b = blockIdx.x;
  int tid = threadIdx.x;
  int lane = tid & 31, w = tid >> 5;

  for (int idx = tid; idx < EMBED * EMBED; idx += 256)
    Uh[(idx >> 8) * 272 + (idx & 255)] = (_Float16)U[idx];
  for (int idx = tid; idx < NB * EMBED; idx += 256) kV[idx] = keyV[idx];
  pals[tid] = prelu_a[tid];
  for (int k = 0; k < 32; ++k) {          // state0 = keys; pad rows 20..31 = 0
    float v = (k < NB) ? embed[(KEY0 + k) * EMBED + tid] : 0.f;
    sF[k * 256 + tid] = v;
    sH[k * 272 + tid] = (_Float16)v;
  }
  __syncthreads();

  for (int t = 0; t < RN; ++t) {
    int bt = b * RN + t;
    xls[tid]  = hist[bt * EMBED + tid];
    xwls[tid] = XW[bt * EMBED + tid];
    __syncthreads();

    // gate[k] = sigmoid(state[k].x + XK)
    for (int k = w; k < NB; k += 8) {
      float p = 0.f;
      for (int j = 0; j < 8; ++j) {
        int e = lane + 32 * j;
        p += sF[k * 256 + e] * xls[e];
      }
      p = wave_reduce(p);
      if (lane == 0) {
        float z = p + XK[bt * NB + k];
        gate[k] = 1.f / (1.f + __expf(-z));
      }
    }

    // cand_mat = state @ U^T  via v_wmma_f32_16x16x32_f16
    {
      int mtile = w & 1;
      int ng    = w >> 1;
      v8f acc[4];
#pragma unroll
      for (int i = 0; i < 4; ++i)
#pragma unroll
        for (int r = 0; r < 8; ++r) acc[i][r] = 0.f;
      for (int kb = 0; kb < 8; ++kb) {
        v16h a = load_a_frag(sH, 272, mtile * 16, kb * 32, lane);
#pragma unroll
        for (int i = 0; i < 4; ++i) {
          v16h bf = load_b_frag(Uh, 272, (ng * 4 + i) * 16, kb * 32, lane);
          acc[i] = __builtin_amdgcn_wmma_f32_16x16x32_f16(
              false, a, false, bf, (short)0, acc[i], false, false);
        }
      }
      int rowhi = (lane >> 4) << 3;
#pragma unroll
      for (int i = 0; i < 4; ++i) {
        int n = (ng * 4 + i) * 16 + (lane & 15);
#pragma unroll
        for (int r = 0; r < 8; ++r)
          cF[(mtile * 16 + r + rowhi) * 256 + n] = acc[i][r];
      }
    }
    __syncthreads();

    // s = state + gate * prelu(cand + xW + keyV)
    for (int k = 0; k < NB; ++k) {
      int e = tid;
      float c = cF[k * 256 + e] + xwls[e] + kV[k * 256 + e];
      c = c > 0.f ? c : pals[e] * c;
      cF[k * 256 + e] = sF[k * 256 + e] + gate[k] * c;
    }
    __syncthreads();

    // row norms
    for (int k = w; k < NB; k += 8) {
      float p = 0.f;
      for (int j = 0; j < 8; ++j) {
        float s = cF[k * 256 + lane + 32 * j];
        p += s * s;
      }
      p = wave_reduce(p);
      if (lane == 0) norms[k] = sqrtf(p) + 1e-8f;
    }
    __syncthreads();

    // state = where(s>0, s, 1.0) / norm
    for (int k = 0; k < NB; ++k) {
      int e = tid;
      float s = cF[k * 256 + e];
      float v = (s > 0.f ? s : 1.f) / norms[k];
      sF[k * 256 + e] = v;
      sH[k * 272 + e] = (_Float16)v;
    }
    __syncthreads();
  }

  for (int k = 0; k < NB; ++k)
    stateOut[(b * NB + k) * EMBED + tid] = sF[k * 256 + tid];
}

// ---------------- attention + u ----------------

__global__ void attn_kernel(const float* stateOut, const float* qenc,
                            _Float16* u16) {
  __shared__ float qls[EMBED];
  __shared__ float logit[NB];
  __shared__ float attn[NB];
  int bq = blockIdx.x;
  int b  = bq / QQ;
  int tid = threadIdx.x, lane = tid & 31, w = tid >> 5;
  qls[tid] = qenc[bq * EMBED + tid];
  __syncthreads();
  for (int k = w; k < NB; k += 8) {
    float p = 0.f;
    for (int j = 0; j < 8; ++j) {
      int e = lane + 32 * j;
      p += stateOut[(b * NB + k) * EMBED + e] * qls[e];
    }
    p = wave_reduce(p);
    if (lane == 0) logit[k] = p;
  }
  __syncthreads();
  if (tid == 0) {
    float m = logit[0];
    for (int k = 1; k < NB; ++k) m = fmaxf(m, logit[k]);
    float s = 0.f;
    for (int k = 0; k < NB; ++k) { float e_ = __expf(logit[k] - m); attn[k] = e_; s += e_; }
    float inv = 1.f / s;
    for (int k = 0; k < NB; ++k) attn[k] *= inv;
  }
  __syncthreads();
  float u = 0.f;
  for (int k = 0; k < NB; ++k)
    u += attn[k] * stateOut[(b * NB + k) * EMBED + tid];
  u16[bq * EMBED + tid] = (_Float16)u;
}

// ---------------- hin = prelu(q_enc + u @ H^T)  (f16 out) ----------------
// grid (20, 2), tile 64M x 128N, K=256

__global__ void hgemm_kernel(const _Float16* u16, const float* H,
                             const float* qenc, const float* prelu_a,
                             _Float16* hin16) {
  extern __shared__ char smem[];
  _Float16* Ah = (_Float16*)smem;            // 64 x 272
  _Float16* Bh = (_Float16*)(smem + 34816);  // 128 x 272
  int m0 = blockIdx.x * 64;
  int n0 = blockIdx.y * 128;
  int tid = threadIdx.x, lane = tid & 31, w = tid >> 5;

  for (int idx = tid; idx < 64 * 32; idx += 256) {
    int r = idx >> 5, c8 = (idx & 31) * 8;
    *(v8h*)(Ah + r * 272 + c8) = *(const v8h*)(u16 + (m0 + r) * EMBED + c8);
  }
  for (int idx = tid; idx < 128 * EMBED; idx += 256) {
    int r = idx >> 8, c = idx & 255;
    Bh[r * 272 + c] = (_Float16)H[(n0 + r) * EMBED + c];
  }
  __syncthreads();

  int msub = w & 3, nh = w >> 2;
  v8f acc[4];
#pragma unroll
  for (int i = 0; i < 4; ++i)
#pragma unroll
    for (int r = 0; r < 8; ++r) acc[i][r] = 0.f;
  for (int kb = 0; kb < 8; ++kb) {
    v16h a = load_a_frag(Ah, 272, msub * 16, kb * 32, lane);
#pragma unroll
    for (int i = 0; i < 4; ++i) {
      v16h bf = load_b_frag(Bh, 272, (nh * 4 + i) * 16, kb * 32, lane);
      acc[i] = __builtin_amdgcn_wmma_f32_16x16x32_f16(
          false, a, false, bf, (short)0, acc[i], false, false);
    }
  }
  int rowhi = (lane >> 4) << 3;
#pragma unroll
  for (int i = 0; i < 4; ++i) {
    int n = n0 + (nh * 4 + i) * 16 + (lane & 15);
    float pa = prelu_a[n];
#pragma unroll
    for (int r = 0; r < 8; ++r) {
      int m = m0 + msub * 16 + r + rowhi;
      float z = acc[i][r] + qenc[m * EMBED + n];
      z = z > 0.f ? z : pa * z;
      hin16[m * EMBED + n] = (_Float16)z;
    }
  }
}

// ---------------- y = hin @ R^T  (the big one) ----------------
// grid (20, 125), tile 64M x 160N, K=256; 20000 = 125*160 -> no tail.
// Tile staging via Tensor Data Mover (async, TENSORcnt-tracked); the dynamic
// LDS region starts at offset 0 (no static __shared__ here), so the LDS byte
// addresses of Ah/Bh are 0 and 34816.

__global__ void rgemm_kernel(const _Float16* hin16, const _Float16* Rh,
                             float* y) {
  extern __shared__ char smem[];
  _Float16* Ah = (_Float16*)smem;            // 64 x 272  @ LDS byte 0
  _Float16* Bh = (_Float16*)(smem + 34816);  // 160 x 272 @ LDS byte 34816
  int m0 = blockIdx.x * 64;
  int n0 = blockIdx.y * 160;
  int tid = threadIdx.x, lane = tid & 31, w = tid >> 5;

  if (w == 0) {   // TDM is a wave op (EXEC-ignored): issue from one wave only
    tdm_load_rows(hin16 + (size_t)m0 * EMBED, 0u, 64);
    tdm_load_rows(Rh + (size_t)n0 * EMBED, 34816u, 160);
    __builtin_amdgcn_s_wait_tensorcnt(0);
  }
  __syncthreads();

  int msub = w & 3, nh = w >> 2;      // 4 Msub x (2 halves of 5 Nsub)
  v8f acc[5];
#pragma unroll
  for (int i = 0; i < 5; ++i)
#pragma unroll
    for (int r = 0; r < 8; ++r) acc[i][r] = 0.f;
  for (int kb = 0; kb < 8; ++kb) {
    v16h a = load_a_frag(Ah, 272, msub * 16, kb * 32, lane);
#pragma unroll
    for (int i = 0; i < 5; ++i) {
      v16h bf = load_b_frag(Bh, 272, (nh * 5 + i) * 16, kb * 32, lane);
      acc[i] = __builtin_amdgcn_wmma_f32_16x16x32_f16(
          false, a, false, bf, (short)0, acc[i], false, false);
    }
  }
  int rowhi = (lane >> 4) << 3;
#pragma unroll
  for (int i = 0; i < 5; ++i) {
    int n = n0 + (nh * 5 + i) * 16 + (lane & 15);
#pragma unroll
    for (int r = 0; r < 8; ++r) {
      int m = m0 + msub * 16 + r + rowhi;
      y[m * ROUT + n] = acc[i][r];
    }
  }
}

// ---------------- launch ----------------

extern "C" void kernel_launch(void* const* d_in, const int* in_sizes, int n_in,
                              void* d_out, int out_size, void* d_ws, size_t ws_size,
                              hipStream_t stream) {
  const int*   qa    = (const int*)d_in[0];
  const int*   fr    = (const int*)d_in[1];
  const float* embed = (const float*)d_in[2];
  const float* pa    = (const float*)d_in[3];
  const float* smask = (const float*)d_in[4];
  const float* qmask = (const float*)d_in[5];
  const float* U     = (const float*)d_in[6];
  const float* V     = (const float*)d_in[7];
  const float* W     = (const float*)d_in[8];
  const float* bias  = (const float*)d_in[9];
  const float* H     = (const float*)d_in[10];
  const float* R     = (const float*)d_in[11];
  float* y = (float*)d_out;

  char* ws = (char*)d_ws;
  float*    histb  = (float*)(ws + 0);          // 640*256 f32
  float*    qenc   = (float*)(ws + 655360);     // 1280*256 f32
  float*    XW     = (float*)(ws + 1966080);    // 640*256 f32
  float*    XK     = (float*)(ws + 2621440);    // 640*20 f32
  float*    keyV   = (float*)(ws + 2672640);    // 20*256 f32
  float*    stateO = (float*)(ws + 2693120);    // 64*20*256 f32
  _Float16* u16    = (_Float16*)(ws + 4003840); // 1280*256 f16
  _Float16* hin16  = (_Float16*)(ws + 4659200); // 1280*256 f16
  _Float16* Rh     = (_Float16*)(ws + 5314560); // 20000*256 f16

  hist_kernel<<<640, 256, 0, stream>>>(fr, embed, smask, histb);
  qenc_kernel<<<1280, 256, 0, stream>>>(qa, embed, qmask, qenc);
  keyv_kernel<<<20, 256, 0, stream>>>(embed, V, keyV);
  xw_kernel<<<640, 256, 0, stream>>>(histb, W, bias, XW);
  xk_kernel<<<640, 256, 0, stream>>>(histb, embed, XK);
  cvt_r_kernel<<<5000, 256, 0, stream>>>(R, Rh);   // 20000*256 / (256*4)
  scan_kernel<<<64, 256, 246016, stream>>>(embed, U, pa, histb, XW, XK, keyV, stateO);
  attn_kernel<<<1280, 256, 0, stream>>>(stateO, qenc, u16);
  hgemm_kernel<<<dim3(20, 2), 256, 104448, stream>>>(u16, H, qenc, pa, hin16);
  rgemm_kernel<<<dim3(20, 125), 256, 121856, stream>>>(hin16, Rh, y);
}
